// HumanBotNet_40269613367304
// MI455X (gfx1250) — matrix operand order repaired and checked by reading, not compiled, not run on hardware
//
#include <hip/hip_runtime.h>
#include <hip/hip_bf16.h>

// ---------------------------------------------------------------------------
// HumanBotNet (Catan policy/value net) forward pass for MI455X (gfx1250).
// Dense layers -> V_WMMA_F32_16X16X32_BF16 (fp32 accumulate).
// LDS tiles are stored PRE-SWIZZLED into WMMA register layout so each lane
// loads its whole 16-elem bf16 fragment with two ds_load_b128.
// segment_sum via hardware f32 global atomics; mish via fast exp identity.
// ---------------------------------------------------------------------------

#define BB   2048
#define NN   54
#define EE   144
#define NFD  18
#define EFD  5
#define FFD  115
#define AD   397
#define HD   48
#define GD   80
#define TD   80
#define PHD  80
#define SHD  32
#define VHD  64
#define NEGV (-1000000000.0f)

typedef __attribute__((ext_vector_type(16))) __bf16 v16bf;
typedef __attribute__((ext_vector_type(8)))  float  v8f;

__device__ __forceinline__ unsigned short f2bf(float f) {
  union { float f; unsigned int u; } x; x.f = f;
  unsigned int r = x.u + 0x7fffu + ((x.u >> 16) & 1u);   // round-to-nearest-even
  return (unsigned short)(r >> 16);
}

// mish(x) = x * tanh(softplus(x)) = x * ((1+e^x)^2 - 1) / ((1+e^x)^2 + 1)
__device__ __forceinline__ float mishf(float x) {
  float s = __expf(fminf(x, 30.f));
  float u = 1.f + s;
  float n = u * u;
  return x * __fdividef(n - 1.f, n + 1.f);
}

// ---- WMMA fragment layout swizzles (wave32, 16x16x32 bf16) ----------------
// A (16xK): lane = m + 16*((k>>3)&1), elem = (k&7) + 8*((k>>4)&1), chunk k>>5.
// B (Kx16): lane = n + 16*((k>>3)&1), elem = (k&7) + 8*((k>>4)&1), chunk k>>5; n-tiles of 16.
// Fragment f stored at [(chunkBase + lane)*16 .. +15] -> 32B contiguous.
__device__ __forceinline__ int swzA(int m, int k) {
  return (((k >> 5) * 32) + m + 16 * ((k >> 3) & 1)) * 16 + ((k & 7) + 8 * ((k >> 4) & 1));
}
__device__ __forceinline__ int swzB(int k, int n, int kChunks) {
  return ((((n >> 4) * kChunks + (k >> 5)) * 32) + (n & 15) + 16 * ((k >> 3) & 1)) * 16 + ((k & 7) + 8 * ((k >> 4) & 1));
}

__device__ __forceinline__ v16bf load_frag(const unsigned short* p) {
  union { uint4 q[2]; v16bf v; } u;
  u.q[0] = ((const uint4*)p)[0];
  u.q[1] = ((const uint4*)p)[1];
  return u.v;
}

#define WMMA_BF16(af, bf_, acc) \
  __builtin_amdgcn_wmma_f32_16x16x32_bf16(false, (af), false, (bf_), (short)0, (acc), false, false)

// ------------------------- generic GEMM: C = act(A @ W + b) -----------------
// A: f32 [M][K] row-major.  W: f32 [K][N] row-major.  grid.y tiles N by 16.
template <int KP, int ACT>
__global__ void __launch_bounds__(128)
gemm_kernel(const float* __restrict__ A, const float* __restrict__ W,
            const float* __restrict__ bias, float* __restrict__ C,
            long long M, int K, int N) {
  constexpr int KC = KP / 32;
  __shared__ alignas(16) unsigned short wS[KP * 16];
  __shared__ float bS[16];
  __shared__ alignas(16) unsigned short xS[4][16 * KP];

  const int t  = threadIdx.x;
  const int n0 = blockIdx.y * 16;
  for (int i = t; i < KP * 16; i += 128) {
    int k = i >> 4, nl = i & 15, n = n0 + nl;
    wS[swzB(k, nl, KC)] = f2bf((k < K && n < N) ? W[(long long)k * N + n] : 0.f);
  }
  if (t < 16) bS[t] = (n0 + t < N) ? bias[n0 + t] : 0.f;

  const int wave = t >> 5, lane = t & 31;
  const long long row0 = ((long long)blockIdx.x * 4 + wave) * 16;
  unsigned short* xs = xS[wave];
  for (int i = lane; i < 16 * KP; i += 32) {
    int m = i / KP, k = i % KP;
    long long r = row0 + m;
    xs[swzA(m, k)] = f2bf((r < M && k < K) ? A[r * (long long)K + k] : 0.f);
  }
  __syncthreads();

  const int half = lane >> 4, lm = lane & 15;
  const int n = n0 + lm;
  v8f acc;
#pragma unroll
  for (int v = 0; v < 8; ++v) acc[v] = bS[lm];
#pragma unroll
  for (int kc = 0; kc < KC; ++kc) {
    v16bf af  = load_frag(xs + (kc * 32 + lane) * 16);
    v16bf bf_ = load_frag(wS + (kc * 32 + lane) * 16);
    acc = WMMA_BF16(af, bf_, acc);
  }
#pragma unroll
  for (int v = 0; v < 8; ++v) {
    long long r = row0 + v + 8 * half;
    if (r < M && n < N) {
      float val = acc[v];
      if (ACT) val = mishf(val);
      C[r * (long long)N + n] = val;
    }
  }
}

// ------------------- GNN message kernel (fused gather + MLP + scatter) ------
// per edge row: x = [h[src](48) | h[dst](48) | ef@We+be (48)] (K=144, pad 160)
// m = mish(W2 @ mish(W1 @ x + b1) + b2); agg[dst] += m  (f32 global atomics)
__global__ void __launch_bounds__(128)
gnn_msg_kernel(const float* __restrict__ hf, const float* __restrict__ ef,
               const int* __restrict__ ei,
               const float* __restrict__ We, const float* __restrict__ be,
               const float* __restrict__ W1, const float* __restrict__ b1,
               const float* __restrict__ W2, const float* __restrict__ b2,
               float* __restrict__ agg) {
  __shared__ float weS[EFD * HD];
  __shared__ float beS[HD], b1S[HD], b2S[HD];
  __shared__ alignas(16) unsigned short w1S[160 * HD];
  __shared__ alignas(16) unsigned short w2S[64 * HD];
  __shared__ alignas(16) unsigned short xS[4][16 * 160];
  __shared__ alignas(16) unsigned short yS[4][16 * 64];
  __shared__ int srcS[4][16];
  __shared__ int dstS[4][16];

  const int t = threadIdx.x;
  for (int i = t; i < EFD * HD; i += 128) weS[i] = We[i];
  for (int i = t; i < HD; i += 128) { beS[i] = be[i]; b1S[i] = b1[i]; b2S[i] = b2[i]; }
  for (int i = t; i < 160 * HD; i += 128) {
    int k = i / HD, n = i % HD;
    w1S[swzB(k, n, 5)] = f2bf(k < 144 ? W1[k * HD + n] : 0.f);
  }
  for (int i = t; i < 64 * HD; i += 128) {
    int k = i / HD, n = i % HD;
    w2S[swzB(k, n, 2)] = f2bf(k < 48 ? W2[k * HD + n] : 0.f);
  }

  const int wave = t >> 5, lane = t & 31;
  const long long row0 = ((long long)blockIdx.x * 4 + wave) * 16;  // B*E % 64 == 0
  if (lane < 16) {
    long long eg = row0 + lane;
    int b  = (int)(eg / EE);
    int el = (int)(eg % EE);
    srcS[wave][lane] = ei[el] + b * NN;
    dstS[wave][lane] = ei[EE + el] + b * NN;
  }
  __syncthreads();

  unsigned short* xs = xS[wave];
  for (int i = lane; i < 16 * HD; i += 32) {
    int m = i / HD, j = i % HD;
    xs[swzA(m, j)]      = f2bf(hf[(long long)srcS[wave][m] * HD + j]);
    xs[swzA(m, 48 + j)] = f2bf(hf[(long long)dstS[wave][m] * HD + j]);
    long long eg = row0 + m;
    float v = beS[j];
#pragma unroll
    for (int k = 0; k < EFD; ++k) v += ef[eg * EFD + k] * weS[k * HD + j];
    xs[swzA(m, 96 + j)] = f2bf(v);
  }
  for (int i = lane; i < 16 * 16; i += 32) xs[swzA(i >> 4, 144 + (i & 15))] = 0;
  for (int i = lane; i < 16 * 16; i += 32) yS[wave][swzA(i >> 4, 48 + (i & 15))] = 0;
  __syncthreads();

  const int half = lane >> 4, lm = lane & 15;
  v16bf ax[5];
#pragma unroll
  for (int kc = 0; kc < 5; ++kc) ax[kc] = load_frag(xs + (kc * 32 + lane) * 16);
#pragma unroll
  for (int nc = 0; nc < 3; ++nc) {                       // 144 -> 48, mish
    int n = nc * 16 + lm;
    v8f acc;
#pragma unroll
    for (int v = 0; v < 8; ++v) acc[v] = b1S[n];
#pragma unroll
    for (int kc = 0; kc < 5; ++kc) {
      v16bf bf_ = load_frag(w1S + ((nc * 5 + kc) * 32 + lane) * 16);
      acc = WMMA_BF16(ax[kc], bf_, acc);
    }
#pragma unroll
    for (int v = 0; v < 8; ++v)
      yS[wave][swzA(v + 8 * half, n)] = f2bf(mishf(acc[v]));
  }
  __syncthreads();
  v16bf ay[2];
#pragma unroll
  for (int kc = 0; kc < 2; ++kc) ay[kc] = load_frag(&yS[wave][0] + (kc * 32 + lane) * 16);
#pragma unroll
  for (int nc = 0; nc < 3; ++nc) {                       // 48 -> 48, mish + scatter
    int n = nc * 16 + lm;
    v8f acc;
#pragma unroll
    for (int v = 0; v < 8; ++v) acc[v] = b2S[n];
#pragma unroll
    for (int kc = 0; kc < 2; ++kc) {
      v16bf bf_ = load_frag(w2S + ((nc * 2 + kc) * 32 + lane) * 16);
      acc = WMMA_BF16(ay[kc], bf_, acc);
    }
#pragma unroll
    for (int v = 0; v < 8; ++v) {
      int m = v + 8 * half;
      float val = mishf(acc[v]);
      __hip_atomic_fetch_add(&agg[(long long)dstS[wave][m] * HD + n], val,
                             __ATOMIC_RELAXED, __HIP_MEMORY_SCOPE_AGENT);
    }
  }
}

// ---------------- GNN update kernel (concat + MLP + residual layernorm) -----
__global__ void __launch_bounds__(128)
gnn_update_kernel(float* __restrict__ hf, const float* __restrict__ agg,
                  const float* __restrict__ U1, const float* __restrict__ bu1,
                  const float* __restrict__ U2, const float* __restrict__ bu2,
                  const float* __restrict__ lnG, const float* __restrict__ lnB) {
  __shared__ alignas(16) unsigned short w1S[96 * HD];
  __shared__ alignas(16) unsigned short w2S[64 * HD];
  __shared__ float b1S[HD], b2S[HD];
  __shared__ alignas(16) unsigned short xS[4][16 * 96];
  __shared__ alignas(16) unsigned short yS[4][16 * 64];
  __shared__ float uS[4][16 * HD];

  const int t = threadIdx.x;
  for (int i = t; i < 96 * HD; i += 128) {
    int k = i / HD, n = i % HD;
    w1S[swzB(k, n, 3)] = f2bf(U1[k * HD + n]);
  }
  for (int i = t; i < 64 * HD; i += 128) {
    int k = i / HD, n = i % HD;
    w2S[swzB(k, n, 2)] = f2bf(k < 48 ? U2[k * HD + n] : 0.f);
  }
  for (int i = t; i < HD; i += 128) { b1S[i] = bu1[i]; b2S[i] = bu2[i]; }

  const int wave = t >> 5, lane = t & 31;
  const long long row0 = ((long long)blockIdx.x * 4 + wave) * 16;  // B*N % 64 == 0
  unsigned short* xs = xS[wave];
  for (int i = lane; i < 16 * 96; i += 32) {
    int m = i / 96, c = i % 96;
    long long r = row0 + m;
    float v = (c < 48) ? hf[r * HD + c] : agg[r * HD + (c - 48)];
    xs[swzA(m, c)] = f2bf(v);
  }
  for (int i = lane; i < 16 * 16; i += 32) yS[wave][swzA(i >> 4, 48 + (i & 15))] = 0;
  __syncthreads();

  const int half = lane >> 4, lm = lane & 15;
  v16bf ax[3];
#pragma unroll
  for (int kc = 0; kc < 3; ++kc) ax[kc] = load_frag(xs + (kc * 32 + lane) * 16);
#pragma unroll
  for (int nc = 0; nc < 3; ++nc) {                       // 96 -> 48, mish
    int n = nc * 16 + lm;
    v8f acc;
#pragma unroll
    for (int v = 0; v < 8; ++v) acc[v] = b1S[n];
#pragma unroll
    for (int kc = 0; kc < 3; ++kc) {
      v16bf bf_ = load_frag(w1S + ((nc * 3 + kc) * 32 + lane) * 16);
      acc = WMMA_BF16(ax[kc], bf_, acc);
    }
#pragma unroll
    for (int v = 0; v < 8; ++v)
      yS[wave][swzA(v + 8 * half, n)] = f2bf(mishf(acc[v]));
  }
  __syncthreads();
  v16bf ay[2];
#pragma unroll
  for (int kc = 0; kc < 2; ++kc) ay[kc] = load_frag(&yS[wave][0] + (kc * 32 + lane) * 16);
#pragma unroll
  for (int nc = 0; nc < 3; ++nc) {                       // 48 -> 48, no act
    int n = nc * 16 + lm;
    v8f acc;
#pragma unroll
    for (int v = 0; v < 8; ++v) acc[v] = b2S[n];
#pragma unroll
    for (int kc = 0; kc < 2; ++kc) {
      v16bf bf_ = load_frag(w2S + ((nc * 2 + kc) * 32 + lane) * 16);
      acc = WMMA_BF16(ay[kc], bf_, acc);
    }
#pragma unroll
    for (int v = 0; v < 8; ++v)
      uS[wave][(v + 8 * half) * HD + n] = acc[v];
  }
  __syncthreads();
  if (lane < 16) {                                       // layernorm(hf + u)
    long long r = row0 + lane;
    float mean = 0.f;
    for (int j = 0; j < HD; ++j) mean += hf[r * HD + j] + uS[wave][lane * HD + j];
    mean *= (1.f / HD);
    float var = 0.f;
    for (int j = 0; j < HD; ++j) {
      float d = hf[r * HD + j] + uS[wave][lane * HD + j] - mean;
      var += d * d;
    }
    var *= (1.f / HD);
    float inv = rsqrtf(var + 1e-5f);
    for (int j = 0; j < HD; ++j) {
      float v = (hf[r * HD + j] + uS[wave][lane * HD + j] - mean) * inv * lnG[j] + lnB[j];
      hf[r * HD + j] = v;
    }
  }
}

// ---------------- policy-head GEMM with fused context gather ----------------
// MODE 0: sett/city   row r: b=r/54;  x=[tn[b](80) | nn[r](48)]            K=128
// MODE 1: road        row r: b=r/72;  x=[tn | nn[b,rp0] | nn[b,rp1]]       K=176
// MODE 2: robber      row r: b=r/19;  x=[tn | mean6(nn[b,tile])]           K=128
// N = 32 (SH), mish activation.
template <int KP, int MODE>
__global__ void __launch_bounds__(128)
head_gemm_kernel(const float* __restrict__ tn, const float* __restrict__ nn,
                 const int* __restrict__ aux,
                 const float* __restrict__ W, const float* __restrict__ bias,
                 float* __restrict__ Hout, long long R, int K) {
  constexpr int KC = KP / 32;
  __shared__ alignas(16) unsigned short wS[KP * 32];
  __shared__ float bS[32];
  __shared__ alignas(16) unsigned short xS[4][16 * KP];

  const int t = threadIdx.x;
  for (int i = t; i < KP * 32; i += 128) {
    int k = i >> 5, n = i & 31;
    wS[swzB(k, n, KC)] = f2bf(k < K ? W[k * 32 + n] : 0.f);
  }
  if (t < 32) bS[t] = bias[t];

  const int wave = t >> 5, lane = t & 31;
  const long long row0 = ((long long)blockIdx.x * 4 + wave) * 16;
  unsigned short* xs = xS[wave];
  for (int i = lane; i < 16 * KP; i += 32) {
    int m = i / KP, c = i % KP;
    long long r = row0 + m;
    float v = 0.f;
    if (r < R) {
      if (MODE == 0) {
        long long b = r / NN;
        if (c < TD)       v = tn[b * TD + c];
        else if (c < 128) v = nn[r * HD + (c - TD)];
      } else if (MODE == 1) {
        long long b = r / 72; int j = (int)(r % 72);
        if (c < TD)       v = tn[b * TD + c];
        else if (c < 128) v = nn[(b * NN + aux[2 * j]) * HD + (c - TD)];
        else if (c < 176) v = nn[(b * NN + aux[2 * j + 1]) * HD + (c - 128)];
      } else {
        long long b = r / 19; int tt = (int)(r % 19);
        if (c < TD)       v = tn[b * TD + c];
        else if (c < 128) {
          float s = 0.f;
#pragma unroll
          for (int q = 0; q < 6; ++q) s += nn[(b * NN + aux[tt * 6 + q]) * HD + (c - TD)];
          v = s * (1.f / 6.f);
        }
      }
    }
    xs[swzA(m, c)] = f2bf(v);
  }
  __syncthreads();

  const int half = lane >> 4, lm = lane & 15;
  v16bf ax[KC];
#pragma unroll
  for (int kc = 0; kc < KC; ++kc) ax[kc] = load_frag(xs + (kc * 32 + lane) * 16);
#pragma unroll
  for (int nc = 0; nc < 2; ++nc) {
    int n = nc * 16 + lm;
    v8f acc;
#pragma unroll
    for (int v = 0; v < 8; ++v) acc[v] = bS[n];
#pragma unroll
    for (int kc = 0; kc < KC; ++kc) {
      v16bf bf_ = load_frag(wS + ((nc * KC + kc) * 32 + lane) * 16);
      acc = WMMA_BF16(ax[kc], bf_, acc);
    }
#pragma unroll
    for (int v = 0; v < 8; ++v) {
      long long r = row0 + v + 8 * half;
      if (r < R) Hout[r * 32 + n] = mishf(acc[v]);
    }
  }
}

// ------------------------------- small scalar kernels -----------------------
__global__ void fill_kernel(float* p, long long n, float v) {
  long long i = (long long)blockIdx.x * blockDim.x + threadIdx.x;
  if (i < n) p[i] = v;
}

__global__ void pool_kernel(const float* __restrict__ hf, float* __restrict__ pooled) {
  int b = blockIdx.x, j = threadIdx.x;
  if (j >= HD) return;
  float s = 0.f, mx = -3.4e38f;
  for (int i = 0; i < NN; ++i) {
    float v = hf[((long long)b * NN + i) * HD + j];
    s += v; mx = fmaxf(mx, v);
  }
  pooled[(long long)b * 96 + j] = s * (1.f / NN);
  pooled[(long long)b * 96 + 48 + j] = mx;
}

__global__ void concat2_kernel(const float* __restrict__ a, int ca,
                               const float* __restrict__ b, int cb,
                               float* __restrict__ o, long long M) {
  long long i = (long long)blockIdx.x * blockDim.x + threadIdx.x;
  int cc = ca + cb;
  if (i >= M * cc) return;
  long long r = i / cc; int c = (int)(i % cc);
  o[i] = (c < ca) ? a[r * ca + c] : b[r * cb + (c - ca)];
}

__global__ void bnstats_kernel(const float* __restrict__ X, long long M, int C,
                               float* __restrict__ mean, float* __restrict__ var) {
  int c = blockIdx.x;
  __shared__ float s1[128], s2[128];
  float a = 0.f, b = 0.f;
  for (long long i = threadIdx.x; i < M; i += 128) {
    float v = X[i * C + c];
    a += v; b += v * v;
  }
  s1[threadIdx.x] = a; s2[threadIdx.x] = b;
  __syncthreads();
  for (int st = 64; st > 0; st >>= 1) {
    if ((int)threadIdx.x < st) { s1[threadIdx.x] += s1[threadIdx.x + st]; s2[threadIdx.x] += s2[threadIdx.x + st]; }
    __syncthreads();
  }
  if (threadIdx.x == 0) {
    float m = s1[0] / (float)M;
    mean[c] = m;
    var[c] = s2[0] / (float)M - m * m;
  }
}

__global__ void bn_apply_kernel(const float* __restrict__ X, const float* __restrict__ mean,
                                const float* __restrict__ var, const float* __restrict__ g,
                                const float* __restrict__ b, float* __restrict__ Y,
                                long long MC, int C, int act) {
  long long i = (long long)blockIdx.x * blockDim.x + threadIdx.x;
  if (i >= MC) return;
  int c = (int)(i % C);
  float v = (X[i] - mean[c]) * rsqrtf(var[c] + 1e-5f) * g[c] + b[c];
  if (act) v = mishf(v);
  Y[i] = v;
}

__global__ void resmish_kernel(float* __restrict__ x, const float* __restrict__ y, long long n) {
  long long i = (long long)blockIdx.x * blockDim.x + threadIdx.x;
  if (i < n) x[i] = mishf(x[i] + y[i]);
}

__global__ void layernorm_kernel(const float* __restrict__ X, const float* __restrict__ g,
                                 const float* __restrict__ b, float* __restrict__ Y,
                                 long long M, int C) {
  long long r = (long long)blockIdx.x * blockDim.x + threadIdx.x;
  if (r >= M) return;
  const float* x = X + r * C;
  float m = 0.f;
  for (int c = 0; c < C; ++c) m += x[c];
  m /= (float)C;
  float v = 0.f;
  for (int c = 0; c < C; ++c) { float d = x[c] - m; v += d * d; }
  v /= (float)C;
  float inv = rsqrtf(v + 1e-5f);
  for (int c = 0; c < C; ++c) Y[r * C + c] = (x[c] - m) * inv * g[c] + b[c];
}

__global__ void proj_kernel(const float* __restrict__ Hb, const float* __restrict__ W,
                            const float* __restrict__ bias, float* __restrict__ out,
                            long long M, int K, int O) {
  long long i = (long long)blockIdx.x * blockDim.x + threadIdx.x;
  if (i >= M * O) return;
  long long r = i / O; int o = (int)(i % O);
  float acc = bias[o];
  for (int k = 0; k < K; ++k) acc += Hb[r * K + k] * W[k * O + o];
  out[i] = acc;
}

__global__ void assemble_kernel(const float* __restrict__ gl, const float* __restrict__ sett,
                                const float* __restrict__ city, const float* __restrict__ road,
                                const float* __restrict__ rob, const int* __restrict__ mask,
                                float* __restrict__ outm) {
  long long i = (long long)blockIdx.x * blockDim.x + threadIdx.x;
  if (i >= (long long)BB * AD) return;
  int b = (int)(i / AD), a = (int)(i % AD);
  float v;
  if (a < 5)        v = gl[(long long)b * AD + a];
  else if (a < 59)  v = sett[(long long)b * NN + (a - 5)];
  else if (a < 113) v = city[(long long)b * NN + (a - 59)];
  else if (a < 185) v = road[(long long)b * 72 + (a - 113)];
  else if (a < 280) v = rob[(long long)b * 95 + (a - 185)];
  else              v = gl[(long long)b * AD + a];
  outm[i] = mask[i] ? v : NEGV;
}

__global__ void softmax_kernel(const float* __restrict__ m, float* __restrict__ p) {
  int r = blockIdx.x * blockDim.x + threadIdx.x;
  if (r >= BB) return;
  const float* x = m + (long long)r * AD;
  float mx = -3.4e38f;
  for (int a = 0; a < AD; ++a) mx = fmaxf(mx, x[a]);
  float s = 0.f;
  for (int a = 0; a < AD; ++a) s += __expf(x[a] - mx);
  float inv = 1.f / s;
  float* o = p + (long long)r * AD;
  for (int a = 0; a < AD; ++a) o[a] = __expf(x[a] - mx) * inv;
}

// ---------------------------------------------------------------------------
extern "C" void kernel_launch(void* const* d_in, const int* in_sizes, int n_in,
                              void* d_out, int out_size, void* d_ws, size_t ws_size,
                              hipStream_t stream) {
  (void)in_sizes; (void)n_in; (void)out_size; (void)ws_size;
  // ---- inputs (setup_inputs order; params flattened in JAX pytree order:
  //      dict keys sorted alphabetically, lists in order, norms {b,g}, lins {b,w}).
  const float* nf   = (const float*)d_in[0];   // (B,N,18)
  const int*   ei   = (const int*)d_in[1];     // (2,E)
  const float* efx  = (const float*)d_in[2];   // (B,E,5)
  const float* ff   = (const float*)d_in[3];   // (B,115)
  const int*   mask = (const int*)d_in[4];     // (B,397)
  // params: blocks[4]{bn1.b,bn1.g,bn2.b,bn2.g,fc1.b,fc1.w,fc2.b,fc2.w} @5..36
  const float* blk[4][8];
  for (int i = 0; i < 4; ++i)
    for (int j = 0; j < 8; ++j) blk[i][j] = (const float*)d_in[5 + 8 * i + j];
  const float* ep_b = (const float*)d_in[37];  // edge_proj.b
  const float* ep_w = (const float*)d_in[38];  // edge_proj.w (5,48)
  // gnn[3]{ln.b,ln.g,m1.b,m1.w,m2.b,m2.w,u1.b,u1.w,u2.b,u2.w} @39..68
  const float* gnn[3][10];
  for (int i = 0; i < 3; ++i)
    for (int j = 0; j < 10; ++j) gnn[i][j] = (const float*)d_in[39 + 10 * i + j];
  const float* np_b = (const float*)d_in[69];
  const float* np_w = (const float*)d_in[70];  // (18,48)
  const float* o1_b = (const float*)d_in[71];
  const float* o1_w = (const float*)d_in[72];  // (96,80)
  const float* o2_b = (const float*)d_in[73];
  const float* o2_w = (const float*)d_in[74];  // (80,80)
  // pol keys sorted: city1,city2,gf1,gf2,gf_bn,node_ln,rob1,rob2,road1,road2,sett1,sett2,trunk_ln
  const float* city1_b = (const float*)d_in[75];
  const float* city1_w = (const float*)d_in[76];
  const float* city2_b = (const float*)d_in[77];
  const float* city2_w = (const float*)d_in[78];
  const float* gf1_b   = (const float*)d_in[79];
  const float* gf1_w   = (const float*)d_in[80];
  const float* gf2_b   = (const float*)d_in[81];
  const float* gf2_w   = (const float*)d_in[82];
  const float* gfbn_b  = (const float*)d_in[83];
  const float* gfbn_g  = (const float*)d_in[84];
  const float* nln_b   = (const float*)d_in[85];
  const float* nln_g   = (const float*)d_in[86];
  const float* rob1_b  = (const float*)d_in[87];
  const float* rob1_w  = (const float*)d_in[88];
  const float* rob2_b  = (const float*)d_in[89];
  const float* rob2_w  = (const float*)d_in[90];
  const float* road1_b = (const float*)d_in[91];
  const float* road1_w = (const float*)d_in[92];
  const float* road2_b = (const float*)d_in[93];
  const float* road2_w = (const float*)d_in[94];
  const float* sett1_b = (const float*)d_in[95];
  const float* sett1_w = (const float*)d_in[96];
  const float* sett2_b = (const float*)d_in[97];
  const float* sett2_w = (const float*)d_in[98];
  const float* tln_b   = (const float*)d_in[99];
  const float* tln_g   = (const float*)d_in[100];
  const float* ti_b    = (const float*)d_in[101];
  const float* ti_w    = (const float*)d_in[102]; // (195,80)
  const float* tibn_b  = (const float*)d_in[103];
  const float* tibn_g  = (const float*)d_in[104];
  const float* vbn_b   = (const float*)d_in[105];
  const float* vbn_g   = (const float*)d_in[106];
  const float* vf1_b   = (const float*)d_in[107];
  const float* vf1_w   = (const float*)d_in[108]; // (80,64)
  const float* vfo_b   = (const float*)d_in[109];
  const float* vfo_w   = (const float*)d_in[110]; // (64,4)
  const int* rp    = (const int*)d_in[111];       // (72,2)
  const int* tiles = (const int*)d_in[112];       // (19,6)

  // ---- workspace layout (f32) ----
  const long long MN = (long long)BB * NN;   // 110592
  const long long MEdg = (long long)BB * EE; // 294912
  float* w = (float*)d_ws;
  size_t off = 0;
  auto alloc = [&](size_t n) { float* p = w + off; off += n; return p; };
  float* hf     = alloc(MN * HD);
  float* agg    = alloc(MN * HD);
  float* nnbuf  = alloc(MN * HD);
  float* pooled = alloc((size_t)BB * 96);
  float* tmpG   = alloc((size_t)BB * GD);
  float* board  = alloc((size_t)BB * GD);
  float* cat195 = alloc((size_t)BB * (GD + FFD));
  float* x      = alloc((size_t)BB * TD);
  float* y      = alloc((size_t)BB * TD);
  float* y2     = alloc((size_t)BB * TD);
  float* tn     = alloc((size_t)BB * TD);
  float* gfh    = alloc((size_t)BB * PHD);
  float* gl     = alloc((size_t)BB * AD);
  float* headH  = alloc((size_t)BB * 72 * SHD);  // max of head hidden sizes
  float* sett   = alloc((size_t)BB * NN);
  float* city   = alloc((size_t)BB * NN);
  float* road   = alloc((size_t)BB * 72);
  float* rob    = alloc((size_t)BB * 95);
  float* valH   = alloc((size_t)BB * VHD);
  float* mn     = alloc(512);
  float* vr     = alloc(512);

  float* outMasked = (float*)d_out;
  float* outProbs  = outMasked + (size_t)BB * AD;
  float* outValue  = outProbs + (size_t)BB * AD;

  auto cdiv = [](long long a, long long b) { return (unsigned)((a + b - 1) / b); };

  // 1) node projection: h = mish(nf @ Wnp + b)   (K=18 -> KP=32, N=48)
  gemm_kernel<32, 1><<<dim3(cdiv(MN, 64), 3), 128, 0, stream>>>(nf, np_w, np_b, hf, MN, NFD, HD);

  // 2) GNN layers
  for (int L = 0; L < 3; ++L) {
    fill_kernel<<<cdiv(MN * HD, 256), 256, 0, stream>>>(agg, MN * HD, 0.f);
    gnn_msg_kernel<<<cdiv(MEdg, 64), 128, 0, stream>>>(
        hf, efx, ei, ep_w, ep_b,
        gnn[L][3], gnn[L][2], gnn[L][5], gnn[L][4], agg);
    gnn_update_kernel<<<cdiv(MN, 64), 128, 0, stream>>>(
        hf, agg, gnn[L][7], gnn[L][6], gnn[L][9], gnn[L][8], gnn[L][1], gnn[L][0]);
  }

  // 3) pooling + board embedding
  pool_kernel<<<BB, 64, 0, stream>>>(hf, pooled);
  gemm_kernel<96, 1><<<dim3(32, 5), 128, 0, stream>>>(pooled, o1_w, o1_b, tmpG, BB, 96, GD);
  gemm_kernel<96, 0><<<dim3(32, 5), 128, 0, stream>>>(tmpG, o2_w, o2_b, board, BB, GD, GD);

  // 4) trunk
  concat2_kernel<<<cdiv((long long)BB * (GD + FFD), 256), 256, 0, stream>>>(board, GD, ff, FFD, cat195, BB);
  gemm_kernel<224, 0><<<dim3(32, 5), 128, 0, stream>>>(cat195, ti_w, ti_b, x, BB, GD + FFD, TD);
  bnstats_kernel<<<TD, 128, 0, stream>>>(x, BB, TD, mn, vr);
  bn_apply_kernel<<<cdiv((long long)BB * TD, 256), 256, 0, stream>>>(x, mn, vr, tibn_g, tibn_b, x, (long long)BB * TD, TD, 1);
  for (int i = 0; i < 4; ++i) {
    gemm_kernel<96, 0><<<dim3(32, 5), 128, 0, stream>>>(x, blk[i][5], blk[i][4], y, BB, TD, TD);
    bnstats_kernel<<<TD, 128, 0, stream>>>(y, BB, TD, mn, vr);
    bn_apply_kernel<<<cdiv((long long)BB * TD, 256), 256, 0, stream>>>(y, mn, vr, blk[i][1], blk[i][0], y, (long long)BB * TD, TD, 1);
    gemm_kernel<96, 0><<<dim3(32, 5), 128, 0, stream>>>(y, blk[i][7], blk[i][6], y2, BB, TD, TD);
    bnstats_kernel<<<TD, 128, 0, stream>>>(y2, BB, TD, mn, vr);
    bn_apply_kernel<<<cdiv((long long)BB * TD, 256), 256, 0, stream>>>(y2, mn, vr, blk[i][3], blk[i][2], y2, (long long)BB * TD, TD, 0);
    resmish_kernel<<<cdiv((long long)BB * TD, 256), 256, 0, stream>>>(x, y2, (long long)BB * TD);
  }

  // 5) policy heads
  layernorm_kernel<<<cdiv(BB, 256), 256, 0, stream>>>(x, tln_g, tln_b, tn, BB, TD);
  layernorm_kernel<<<cdiv(MN, 256), 256, 0, stream>>>(hf, nln_g, nln_b, nnbuf, MN, HD);
  gemm_kernel<96, 0><<<dim3(32, 5), 128, 0, stream>>>(tn, gf1_w, gf1_b, gfh, BB, TD, PHD);
  bnstats_kernel<<<PHD, 128, 0, stream>>>(gfh, BB, PHD, mn, vr);
  bn_apply_kernel<<<cdiv((long long)BB * PHD, 256), 256, 0, stream>>>(gfh, mn, vr, gfbn_g, gfbn_b, gfh, (long long)BB * PHD, PHD, 1);
  gemm_kernel<96, 0><<<dim3(32, 25), 128, 0, stream>>>(gfh, gf2_w, gf2_b, gl, BB, PHD, AD);

  head_gemm_kernel<128, 0><<<cdiv(MN, 64), 128, 0, stream>>>(tn, nnbuf, nullptr, sett1_w, sett1_b, headH, MN, 128);
  proj_kernel<<<cdiv(MN, 256), 256, 0, stream>>>(headH, sett2_w, sett2_b, sett, MN, SHD, 1);
  head_gemm_kernel<128, 0><<<cdiv(MN, 64), 128, 0, stream>>>(tn, nnbuf, nullptr, city1_w, city1_b, headH, MN, 128);
  proj_kernel<<<cdiv(MN, 256), 256, 0, stream>>>(headH, city2_w, city2_b, city, MN, SHD, 1);
  const long long MR = (long long)BB * 72;
  head_gemm_kernel<192, 1><<<cdiv(MR, 64), 128, 0, stream>>>(tn, nnbuf, rp, road1_w, road1_b, headH, MR, 176);
  proj_kernel<<<cdiv(MR, 256), 256, 0, stream>>>(headH, road2_w, road2_b, road, MR, SHD, 1);
  const long long MT = (long long)BB * 19;
  head_gemm_kernel<128, 2><<<cdiv(MT, 64), 128, 0, stream>>>(tn, nnbuf, tiles, rob1_w, rob1_b, headH, MT, 128);
  proj_kernel<<<cdiv(MT * 5, 256), 256, 0, stream>>>(headH, rob2_w, rob2_b, rob, MT, SHD, 5);

  // 6) assemble logits, mask, softmax
  assemble_kernel<<<cdiv((long long)BB * AD, 256), 256, 0, stream>>>(gl, sett, city, road, rob, mask, outMasked);
  softmax_kernel<<<cdiv(BB, 128), 128, 0, stream>>>(outMasked, outProbs);

  // 7) value head
  gemm_kernel<96, 0><<<dim3(32, 4), 128, 0, stream>>>(x, vf1_w, vf1_b, valH, BB, TD, VHD);
  bnstats_kernel<<<VHD, 128, 0, stream>>>(valH, BB, VHD, mn, vr);
  bn_apply_kernel<<<cdiv((long long)BB * VHD, 256), 256, 0, stream>>>(valH, mn, vr, vbn_g, vbn_b, valH, (long long)BB * VHD, VHD, 1);
  proj_kernel<<<cdiv((long long)BB * 4, 256), 256, 0, stream>>>(valH, vfo_w, vfo_b, outValue, BB, VHD, 4);
}
